// TransformerLayer_34299608826071
// MI455X (gfx1250) — matrix operand order repaired and checked by reading, not compiled, not run on hardware
//
#include <hip/hip_runtime.h>
#include <math.h>
#include <type_traits>

// ---------- types ----------
typedef __attribute__((ext_vector_type(16))) __bf16 v16bf;
typedef __attribute__((ext_vector_type(8)))  float  v8f;
typedef __attribute__((ext_vector_type(4)))  unsigned int u32x4;
typedef __attribute__((ext_vector_type(4)))  float  f32x4;
typedef int i32x4v __attribute__((vector_size(16)));   // matches builtin param pointee

#if defined(__gfx1250__) && __has_builtin(__builtin_amdgcn_global_load_async_to_lds_b128)
#define USE_ASYNC_LDS 1
#else
#define USE_ASYNC_LDS 0
#endif

struct U32x4Pair { u32x4 lo, hi; };

__device__ __forceinline__ v16bf load_frag_pair(const void* p_lo, const void* p_hi) {
  U32x4Pair t;
  t.lo = *reinterpret_cast<const u32x4*>(p_lo);
  t.hi = *reinterpret_cast<const u32x4*>(p_hi);
  return __builtin_bit_cast(v16bf, t);
}
__device__ __forceinline__ v16bf load_frag_contig(const void* p) {
  return load_frag_pair(p, (const char*)p + 16);
}
__device__ __forceinline__ v8f wmma_bf16(v16bf a, v16bf b, v8f c) {
  return __builtin_amdgcn_wmma_f32_16x16x32_bf16(false, a, false, b, (short)0, c, false, false);
}

#if USE_ASYNC_LDS
typedef __attribute__((address_space(1))) i32x4v* gvec_ptr;   // global int4*
typedef __attribute__((address_space(3))) i32x4v* lvec_ptr;   // LDS int4*
__device__ __forceinline__ void async_ld_lds_b128(const void* g, void* l) {
  __builtin_amdgcn_global_load_async_to_lds_b128((gvec_ptr)g, (lvec_ptr)l, 0, 0);
}
__device__ __forceinline__ void wait_async0() {
#if __has_builtin(__builtin_amdgcn_s_wait_asynccnt)
  __builtin_amdgcn_s_wait_asynccnt(0);
#else
  asm volatile("s_wait_asynccnt 0x0" ::: "memory");
#endif
}
#endif

// ---------- weight transpose + fp32->bf16 : W[K][N] -> Wt[N][K] ----------
__global__ __launch_bounds__(256)
void transpose_to_bf16(const float* __restrict__ W, __bf16* __restrict__ Wt, int K, int N)
{
  __shared__ float tile[32][33];
  const int k0 = blockIdx.x * 32, n0 = blockIdx.y * 32;
  const int tx = threadIdx.x & 31, ty = threadIdx.x >> 5;
  #pragma unroll
  for (int i = ty; i < 32; i += 8)
    tile[i][tx] = W[(size_t)(k0 + i) * N + n0 + tx];
  __syncthreads();
  #pragma unroll
  for (int i = ty; i < 32; i += 8)
    Wt[(size_t)(n0 + i) * K + k0 + tx] = (__bf16)tile[tx][i];
}

// ---------- V transpose per head : v[B*S][1024] -> vt[B*H][64][S] (bf16) ----------
__global__ __launch_bounds__(256)
void transpose_v_kernel(const __bf16* __restrict__ v, __bf16* __restrict__ vt)
{
  __shared__ __bf16 tile[32][33];
  const int s0 = blockIdx.x * 32;
  const int y  = blockIdx.y;
  const int d0 = (y & 1) * 32;
  const int bh = y >> 1;            // b*16 + h
  const int b  = bh >> 4, h = bh & 15;
  const int tx = threadIdx.x & 31, ty = threadIdx.x >> 5;
  #pragma unroll
  for (int i = ty; i < 32; i += 8)
    tile[i][tx] = v[(size_t)(b * 2048 + s0 + i) * 1024 + h * 64 + d0 + tx];
  __syncthreads();
  #pragma unroll
  for (int i = ty; i < 32; i += 8)
    vt[((size_t)bh * 64 + d0 + i) * 2048 + s0 + tx] = tile[tx][i];
}

// ---------- LayerNorm fp32 -> bf16, D = 1024, one row per block ----------
__global__ __launch_bounds__(256)
void layernorm_to_bf16(const float* __restrict__ x, const float* __restrict__ g,
                       const float* __restrict__ bb, __bf16* __restrict__ out)
{
  const int row = blockIdx.x, t = threadIdx.x;
  const f32x4 f = reinterpret_cast<const f32x4*>(x + (size_t)row * 1024)[t];
  float s1 = f.x + f.y + f.z + f.w;
  float s2 = f.x * f.x + f.y * f.y + f.z * f.z + f.w * f.w;
  #pragma unroll
  for (int o = 16; o >= 1; o >>= 1) { s1 += __shfl_xor(s1, o); s2 += __shfl_xor(s2, o); }
  __shared__ float r1[8], r2[8];
  const int wid = t >> 5, lane = t & 31;
  if (lane == 0) { r1[wid] = s1; r2[wid] = s2; }
  __syncthreads();
  if (wid == 0) {
    float a = (lane < 8) ? r1[lane] : 0.f;
    float c = (lane < 8) ? r2[lane] : 0.f;
    #pragma unroll
    for (int o = 4; o >= 1; o >>= 1) { a += __shfl_xor(a, o); c += __shfl_xor(c, o); }
    if (lane == 0) { r1[0] = a; r2[0] = c; }
  }
  __syncthreads();
  const float mu = r1[0] * (1.f / 1024.f);
  const float var = r2[0] * (1.f / 1024.f) - mu * mu;
  const float rs = rsqrtf(var + 1e-5f);
  const f32x4 g4 = reinterpret_cast<const f32x4*>(g)[t];
  const f32x4 b4 = reinterpret_cast<const f32x4*>(bb)[t];
  __bf16* orow = out + (size_t)row * 1024 + t * 4;
  orow[0] = (__bf16)((f.x - mu) * rs * g4.x + b4.x);
  orow[1] = (__bf16)((f.y - mu) * rs * g4.y + b4.y);
  orow[2] = (__bf16)((f.z - mu) * rs * g4.z + b4.z);
  orow[3] = (__bf16)((f.w - mu) * rs * g4.w + b4.w);
}

// ---------- tiled WMMA GEMM (double-buffered LDS, async staging if available) ----------
// C[M][N] = A[M][K] * Bt[N][K]^T + bias, with epilogue variants.
enum { EPI_BF16 = 0, EPI_F32_RESID = 1, EPI_BF16_GELU = 2 };

template<int EPI>
__global__ __launch_bounds__(256)
void gemm_bf16_kernel(const __bf16* __restrict__ A, const __bf16* __restrict__ Bt,
                      const float* __restrict__ bias, const float* __restrict__ resid,
                      void* __restrict__ outv, int M, int N, int K)
{
  __shared__ __align__(64) __bf16 As[2][128][32];   // 2 x 8 KB
  __shared__ __align__(64) __bf16 Bs[2][128][32];   // 2 x 8 KB
  const int tid  = threadIdx.x;
  const int lane = tid & 31, wid = tid >> 5;
  const int half = lane >> 4, l16 = lane & 15;
  const int bm = blockIdx.x * 128, bn = blockIdx.y * 128;
  const int wm = (wid >> 1) * 32, wn = (wid & 1) * 64;   // 4x2 wave grid
  // each thread stages 2 slots of A and 2 of B per tile (s = tid, tid+256)
  const int row0 = tid >> 2,           seg0 = (tid & 3) << 3;
  const int row1 = (tid + 256) >> 2,   seg1 = seg0;        // (s&3) identical
  v8f acc[2][4] = {};

#if USE_ASYNC_LDS
  // ---- async global -> LDS pipeline ----
  {
    async_ld_lds_b128(&A [(size_t)(bm + row0) * K + seg0], &As[0][row0][seg0]);
    async_ld_lds_b128(&A [(size_t)(bm + row1) * K + seg1], &As[0][row1][seg1]);
    async_ld_lds_b128(&Bt[(size_t)(bn + row0) * K + seg0], &Bs[0][row0][seg0]);
    async_ld_lds_b128(&Bt[(size_t)(bn + row1) * K + seg1], &Bs[0][row1][seg1]);
    wait_async0();
  }
  __syncthreads();
#else
  // ---- register-staged pipeline ----
  u32x4 ra0, ra1, rb0, rb1;
  ra0 = *reinterpret_cast<const u32x4*>(&A [(size_t)(bm + row0) * K + seg0]);
  ra1 = *reinterpret_cast<const u32x4*>(&A [(size_t)(bm + row1) * K + seg1]);
  rb0 = *reinterpret_cast<const u32x4*>(&Bt[(size_t)(bn + row0) * K + seg0]);
  rb1 = *reinterpret_cast<const u32x4*>(&Bt[(size_t)(bn + row1) * K + seg1]);
  *reinterpret_cast<u32x4*>(&As[0][row0][seg0]) = ra0;
  *reinterpret_cast<u32x4*>(&As[0][row1][seg1]) = ra1;
  *reinterpret_cast<u32x4*>(&Bs[0][row0][seg0]) = rb0;
  *reinterpret_cast<u32x4*>(&Bs[0][row1][seg1]) = rb1;
  __syncthreads();
#endif

  int buf = 0;
  for (int k0 = 0; k0 < K; k0 += 32) {
    const bool has_next = (k0 + 32) < K;
    const int kn = k0 + 32;
#if USE_ASYNC_LDS
    if (has_next) {
      async_ld_lds_b128(&A [(size_t)(bm + row0) * K + kn + seg0], &As[buf ^ 1][row0][seg0]);
      async_ld_lds_b128(&A [(size_t)(bm + row1) * K + kn + seg1], &As[buf ^ 1][row1][seg1]);
      async_ld_lds_b128(&Bt[(size_t)(bn + row0) * K + kn + seg0], &Bs[buf ^ 1][row0][seg0]);
      async_ld_lds_b128(&Bt[(size_t)(bn + row1) * K + kn + seg1], &Bs[buf ^ 1][row1][seg1]);
    }
#else
    if (has_next) {
      ra0 = *reinterpret_cast<const u32x4*>(&A [(size_t)(bm + row0) * K + kn + seg0]);
      ra1 = *reinterpret_cast<const u32x4*>(&A [(size_t)(bm + row1) * K + kn + seg1]);
      rb0 = *reinterpret_cast<const u32x4*>(&Bt[(size_t)(bn + row0) * K + kn + seg0]);
      rb1 = *reinterpret_cast<const u32x4*>(&Bt[(size_t)(bn + row1) * K + kn + seg1]);
    }
#endif
    // ---- compute from buffer `buf` (overlaps the staging above) ----
    v16bf af[2], bf[4];
    #pragma unroll
    for (int mt = 0; mt < 2; ++mt) {
      const int r = wm + mt * 16 + l16;
      // A 16x32 fragment: K = 16h..16h+15 contiguous
      af[mt] = load_frag_contig(&As[buf][r][16 * half]);
    }
    #pragma unroll
    for (int nt = 0; nt < 4; ++nt) {
      const int c = wn + nt * 16 + l16;
      // B 32x16 fragment: K = 16h..16h+15 contiguous
      bf[nt] = load_frag_contig(&Bs[buf][c][16 * half]);
    }
    #pragma unroll
    for (int mt = 0; mt < 2; ++mt)
      #pragma unroll
      for (int nt = 0; nt < 4; ++nt)
        acc[mt][nt] = wmma_bf16(af[mt], bf[nt], acc[mt][nt]);

#if USE_ASYNC_LDS
    if (has_next) wait_async0();
#else
    if (has_next) {
      *reinterpret_cast<u32x4*>(&As[buf ^ 1][row0][seg0]) = ra0;
      *reinterpret_cast<u32x4*>(&As[buf ^ 1][row1][seg1]) = ra1;
      *reinterpret_cast<u32x4*>(&Bs[buf ^ 1][row0][seg0]) = rb0;
      *reinterpret_cast<u32x4*>(&Bs[buf ^ 1][row1][seg1]) = rb1;
    }
#endif
    __syncthreads();
    buf ^= 1;
  }

  #pragma unroll
  for (int mt = 0; mt < 2; ++mt) {
    #pragma unroll
    for (int nt = 0; nt < 4; ++nt) {
      const int cg = bn + wn + nt * 16 + l16;
      const float bv = bias[cg];
      #pragma unroll
      for (int i = 0; i < 8; ++i) {
        const int rg = bm + wm + mt * 16 + i + 8 * half;
        const size_t idx = (size_t)rg * N + cg;
        float v = acc[mt][nt][i] + bv;
        if constexpr (EPI == EPI_BF16) {
          ((__bf16*)outv)[idx] = (__bf16)v;
        } else if constexpr (EPI == EPI_F32_RESID) {
          ((float*)outv)[idx] = v + resid[idx];
        } else {
          v = 0.5f * v * (1.0f + erff(v * 0.70710678118654752f));
          ((__bf16*)outv)[idx] = (__bf16)v;
        }
      }
    }
  }
}

// ---------- causal flash attention, transposed-score trick ----------
// q,k: bf16 [B*S][1024]; vt: bf16 [B*H][64][S]; o: bf16 [B*S][1024]
// K fragments are double-buffered (full-chunk prefetch distance); V fragments
// are loaded at the top of process() — the score WMMAs + softmax cover their
// latency — keeping live VGPRs under the 256 boundary.
struct KFrag { v16bf kf[2][2]; };

__global__ __launch_bounds__(256)
void flash_attn_kernel(const __bf16* __restrict__ q, const __bf16* __restrict__ k,
                       const __bf16* __restrict__ vt, __bf16* __restrict__ o)
{
  constexpr int S = 2048;
  const int lane = threadIdx.x & 31, wid = threadIdx.x >> 5;
  const int half = lane >> 4, l16 = lane & 15;
  const int bx = blockIdx.x;
  const int q0 = (bx & 15) * 128 + wid * 16;   // 16 q-strips of 128 per (b,h)
  const int bh = bx >> 4;
  const int b  = bh >> 4, h = bh & 15;
  const float scale = 0.125f;                  // 1/sqrt(64)

  // Q^T B-fragments (col = query = l16, 16 consecutive d per half), reused all chunks
  const __bf16* qrow = q + (size_t)(b * S + q0 + l16) * 1024 + h * 64;
  v16bf qf[2];
  #pragma unroll
  for (int dc = 0; dc < 2; ++dc)
    qf[dc] = load_frag_contig(qrow + dc * 32 + 16 * half);

  const __bf16* kbase = k + (size_t)(b * S + l16) * 1024 + h * 64;
  const __bf16* vbase[4];
  #pragma unroll
  for (int dc = 0; dc < 4; ++dc)
    vbase[dc] = vt + ((size_t)bh * 64 + dc * 16 + l16) * S + 8 * half;

  v8f oacc[4] = {};
  float m = -__builtin_inff(), l = 0.f;
  const int qg = q0 + l16;

  auto load_k = [&](int kb, KFrag& c) {
    #pragma unroll
    for (int t = 0; t < 2; ++t) {
      const __bf16* krow = kbase + (size_t)(kb + t * 16) * 1024;
      #pragma unroll
      for (int dc = 0; dc < 2; ++dc)
        c.kf[t][dc] = load_frag_contig(krow + dc * 32 + 16 * half);
    }
  };

  auto process = [&](int kb, const KFrag& c, auto masked_tag) {
    constexpr bool MASKED = decltype(masked_tag)::value;
    // issue V fragment loads first: consumed only after scores + softmax
    v16bf vf[4];
    #pragma unroll
    for (int dc = 0; dc < 4; ++dc)
      vf[dc] = load_frag_pair(vbase[dc] + kb, vbase[dc] + kb + 16);
    // S^T tiles: rows = keys, cols = queries  (C-layout: key = kb+t*16+i+8h, q = l16)
    v8f st[2];
    #pragma unroll
    for (int t = 0; t < 2; ++t) {
      v8f cc = {};
      #pragma unroll
      for (int dc = 0; dc < 2; ++dc)
        cc = wmma_bf16(c.kf[t][dc], qf[dc], cc);
      st[t] = cc;
    }
    // online softmax per query (= per lane column, merged across halves)
    float p0[8], p1[8];
    float cmax = -__builtin_inff();
    #pragma unroll
    for (int i = 0; i < 8; ++i) {
      if constexpr (MASKED) {
        const int kg0 = kb + i + 8 * half;
        p0[i] = (kg0      <= qg) ? st[0][i] * scale : -__builtin_inff();
        p1[i] = (kg0 + 16 <= qg) ? st[1][i] * scale : -__builtin_inff();
      } else {
        p0[i] = st[0][i] * scale;
        p1[i] = st[1][i] * scale;
      }
      cmax = fmaxf(cmax, fmaxf(p0[i], p1[i]));
    }
    cmax = fmaxf(cmax, __shfl_xor(cmax, 16));
    const float mnew = fmaxf(m, cmax);
    const float fac = __expf(m - mnew);
    float csum = 0.f;
    v16bf pa;   // P A-fragment: element e <-> key 8h+e (tile0), 8+e <-> key 16+8h+e (tile1)
    #pragma unroll
    for (int i = 0; i < 8; ++i) {
      const float e0 = __expf(p0[i] - mnew);
      const float e1 = __expf(p1[i] - mnew);
      csum += e0 + e1;
      pa[i]     = (__bf16)e0;
      pa[i + 8] = (__bf16)e1;
    }
    csum += __shfl_xor(csum, 16);
    l = l * fac + csum;
    m = mnew;
    // rescale O (rows = q = i+8h): broadcast per-row factors from lane i+8h
    float rowf[8];
    #pragma unroll
    for (int i = 0; i < 8; ++i) rowf[i] = __shfl(fac, 8 * half + i);
    #pragma unroll
    for (int dc = 0; dc < 4; ++dc) {
      #pragma unroll
      for (int i = 0; i < 8; ++i) oacc[dc][i] *= rowf[i];
      oacc[dc] = wmma_bf16(pa, vf[dc], oacc[dc]);
    }
  };

  // 2x-unrolled pipeline over fully-unmasked chunks; the single diagonal chunk
  // kb_m is the pipeline tail with masking enabled.
  const int kb_m = (q0 >> 5) << 5;
  KFrag fA, fB;
  load_k(0, fA);
  int kb = 0;
  for (;;) {
    if (kb >= kb_m) { process(kb, fA, std::true_type{}); break; }
    load_k(kb + 32, fB);
    process(kb, fA, std::false_type{});
    kb += 32;
    if (kb >= kb_m) { process(kb, fB, std::true_type{}); break; }
    load_k(kb + 32, fA);
    process(kb, fB, std::false_type{});
    kb += 32;
  }

  float rowl[8];
  #pragma unroll
  for (int i = 0; i < 8; ++i) rowl[i] = __shfl(l, 8 * half + i);
  #pragma unroll
  for (int dc = 0; dc < 4; ++dc)
    #pragma unroll
    for (int i = 0; i < 8; ++i) {
      const int qr = q0 + i + 8 * half;
      o[(size_t)(b * S + qr) * 1024 + h * 64 + dc * 16 + l16] =
          (__bf16)(oacc[dc][i] / rowl[i]);
    }
}

// ---------- host orchestration ----------
extern "C" void kernel_launch(void* const* d_in, const int* in_sizes, int n_in,
                              void* d_out, int out_size, void* d_ws, size_t ws_size,
                              hipStream_t stream)
{
  (void)in_sizes; (void)n_in; (void)out_size; (void)ws_size;
  constexpr int D = 1024, F = 4096, Bb = 4, S = 2048, M = Bb * S;

  const float* x    = (const float*)d_in[0];
  const float* wq   = (const float*)d_in[1];
  const float* bq   = (const float*)d_in[2];
  const float* wk   = (const float*)d_in[3];
  const float* bk   = (const float*)d_in[4];
  const float* wv   = (const float*)d_in[5];
  const float* bv   = (const float*)d_in[6];
  const float* wo   = (const float*)d_in[7];
  const float* bo   = (const float*)d_in[8];
  const float* w1   = (const float*)d_in[9];
  const float* b1   = (const float*)d_in[10];
  const float* w2   = (const float*)d_in[11];
  const float* b2   = (const float*)d_in[12];
  const float* ln1g = (const float*)d_in[13];
  const float* ln1b = (const float*)d_in[14];
  const float* ln2g = (const float*)d_in[15];
  const float* ln2b = (const float*)d_in[16];

  size_t off = 0;
  auto wsp = [&](size_t bytes) -> void* {
    void* p = (char*)d_ws + off;
    off += (bytes + 255) & ~(size_t)255;
    return p;
  };
  __bf16* wqt  = (__bf16*)wsp((size_t)D * D * 2);
  __bf16* wkt  = (__bf16*)wsp((size_t)D * D * 2);
  __bf16* wvt  = (__bf16*)wsp((size_t)D * D * 2);
  __bf16* wot  = (__bf16*)wsp((size_t)D * D * 2);
  __bf16* w1t  = (__bf16*)wsp((size_t)D * F * 2);
  __bf16* w2t  = (__bf16*)wsp((size_t)F * D * 2);
  __bf16* h1   = (__bf16*)wsp((size_t)M * D * 2);
  __bf16* qb   = (__bf16*)wsp((size_t)M * D * 2);
  __bf16* kb_  = (__bf16*)wsp((size_t)M * D * 2);
  __bf16* vb_  = (__bf16*)wsp((size_t)M * D * 2);
  __bf16* vtb  = (__bf16*)wsp((size_t)M * D * 2);
  __bf16* attn = (__bf16*)wsp((size_t)M * D * 2);
  float*  xres = (float*) wsp((size_t)M * D * 4);
  __bf16* h2   = (__bf16*)wsp((size_t)M * D * 2);
  __bf16* ffh  = (__bf16*)wsp((size_t)M * F * 2);

  // weights -> bf16, transposed to [N][K]
  transpose_to_bf16<<<dim3(D / 32, D / 32), 256, 0, stream>>>(wq, wqt, D, D);
  transpose_to_bf16<<<dim3(D / 32, D / 32), 256, 0, stream>>>(wk, wkt, D, D);
  transpose_to_bf16<<<dim3(D / 32, D / 32), 256, 0, stream>>>(wv, wvt, D, D);
  transpose_to_bf16<<<dim3(D / 32, D / 32), 256, 0, stream>>>(wo, wot, D, D);
  transpose_to_bf16<<<dim3(D / 32, F / 32), 256, 0, stream>>>(w1, w1t, D, F);
  transpose_to_bf16<<<dim3(F / 32, D / 32), 256, 0, stream>>>(w2, w2t, F, D);

  // LN1 -> h1 (bf16)
  layernorm_to_bf16<<<M, 256, 0, stream>>>(x, ln1g, ln1b, h1);

  // QKV projections (bf16 out)
  gemm_bf16_kernel<EPI_BF16><<<dim3(M / 128, D / 128), 256, 0, stream>>>(h1, wqt, bq, nullptr, qb,  M, D, D);
  gemm_bf16_kernel<EPI_BF16><<<dim3(M / 128, D / 128), 256, 0, stream>>>(h1, wkt, bk, nullptr, kb_, M, D, D);
  gemm_bf16_kernel<EPI_BF16><<<dim3(M / 128, D / 128), 256, 0, stream>>>(h1, wvt, bv, nullptr, vb_, M, D, D);

  // V -> per-head transposed [B*H][64][S]
  transpose_v_kernel<<<dim3(S / 32, Bb * 16 * 2), 256, 0, stream>>>(vb_, vtb);

  // causal flash attention
  flash_attn_kernel<<<Bb * 16 * (S / 128), 256, 0, stream>>>(qb, kb_, vtb, attn);

  // O-projection + residual (fp32)
  gemm_bf16_kernel<EPI_F32_RESID><<<dim3(M / 128, D / 128), 256, 0, stream>>>(attn, wot, bo, x, xres, M, D, D);

  // LN2 -> h2 (bf16)
  layernorm_to_bf16<<<M, 256, 0, stream>>>(xres, ln2g, ln2b, h2);

  // FFN1 + exact GELU (bf16)
  gemm_bf16_kernel<EPI_BF16_GELU><<<dim3(M / 128, F / 128), 256, 0, stream>>>(h2, w1t, b1, nullptr, ffh, M, F, D);

  // FFN2 + residual -> d_out (fp32)
  gemm_bf16_kernel<EPI_F32_RESID><<<dim3(M / 128, D / 128), 256, 0, stream>>>(ffh, w2t, b2, xres, (float*)d_out, M, D, F);
}
